// TensorizedPC_81767587381968
// MI455X (gfx1250) — matrix operand order repaired and checked by reading, not compiled, not run on hardware
//
#include <hip/hip_runtime.h>
#include <math.h>

// ---------------------------------------------------------------------------
// Tensorized PC (CP sum-product circuit) forward for MI455X / gfx1250.
// V=1024 vars, B=256 batch, K=128 units. 10 CP layers; layer 1 fuses the
// Gaussian input layer; layers 1..9 use v_wmma_f32_16x16x32_f16; layer 10
// (Ko=1) is a scalar reduction.
// ---------------------------------------------------------------------------

typedef __attribute__((ext_vector_type(16))) _Float16 v16h;
typedef __attribute__((ext_vector_type(8)))  _Float16 v8h;
typedef __attribute__((ext_vector_type(8)))  float    v8f;

static constexpr int   Bsz   = 256;   // batch
static constexpr int   Kd    = 128;   // units per region
static constexpr int   Vn    = 1024;  // num vars
static constexpr int   BT    = 64;    // batch rows per workgroup
static constexpr int   ASTR  = 136;   // f16 stride of A tile rows (16B aligned, bank-rotated)
static constexpr int   WSTR  = 80;    // f16 stride of W chunk rows (16B aligned, bank-rotated)
static constexpr float NLOG2PI_2 = -0.91893853320467274178f; // -0.5*log(2*pi)

template<bool FUSED>
__global__ __launch_bounds__(256)
void pc_cp_layer(const float* __restrict__ x, const float* __restrict__ mu,
                 const float* __restrict__ prev, const float* __restrict__ w,
                 float* __restrict__ out)
{
  __shared__ alignas(16) _Float16 sA[2][BT][ASTR]; // exp(inp - m), f16, both children
  __shared__ alignas(16) _Float16 sW[Kd][WSTR];    // transposed weight chunk [o][i]
  __shared__ alignas(16) float    sM[2][BT];       // per-row max (log-space shift)
  __shared__ alignas(16) float    sPM[256];        // partial-max scratch

  const int tid = threadIdx.x;
  const int f   = blockIdx.y;        // fold
  const int bb  = blockIdx.x * BT;   // batch tile base

  // ---- Stage A for both children: rowmax + exp -> f16 LDS -----------------
  {
    const int rh = tid >> 1;         // 0..127 == (child, row)
    const int a  = rh >> 6;
    const int r  = rh & 63;
    const int h  = tid & 1;          // K half: [h*64, h*64+64)
    float vals[64];
    float pm = -3.0e38f;
    if constexpr (FUSED) {
      // input layer fused: inp = -0.5*(x - mu)^2 - 0.5*log(2*pi)
      const int   v  = 2 * f + a;
      const float xv = x[(size_t)(bb + r) * Vn + v];
      const float4* mu4 = (const float4*)(mu + (size_t)v * Kd + h * 64);
      #pragma unroll
      for (int i = 0; i < 16; ++i) {
        float4 m4 = mu4[i];
        float d0 = xv - m4.x, d1 = xv - m4.y, d2 = xv - m4.z, d3 = xv - m4.w;
        vals[4*i+0] = -0.5f*d0*d0 + NLOG2PI_2;
        vals[4*i+1] = -0.5f*d1*d1 + NLOG2PI_2;
        vals[4*i+2] = -0.5f*d2*d2 + NLOG2PI_2;
        vals[4*i+3] = -0.5f*d3*d3 + NLOG2PI_2;
      }
    } else {
      const float4* p4 =
          (const float4*)(prev + ((size_t)(2*f + a) * Bsz + bb + r) * Kd + h * 64);
      #pragma unroll
      for (int i = 0; i < 16; ++i) {
        float4 t = p4[i];
        vals[4*i+0] = t.x; vals[4*i+1] = t.y; vals[4*i+2] = t.z; vals[4*i+3] = t.w;
      }
    }
    #pragma unroll
    for (int i = 0; i < 64; ++i) pm = fmaxf(pm, vals[i]);
    sPM[tid] = pm;
    __syncthreads();
    const float m = fmaxf(sPM[rh*2], sPM[rh*2 + 1]);
    if (h == 0) sM[a][r] = m;
    #pragma unroll
    for (int i = 0; i < 64; ++i)
      sA[a][r][h*64 + i] = (_Float16)__expf(vals[i] - m);
  }
  __syncthreads();

  // ---- WMMA GEMMs: acc[child][ntile] = exp(inp-m) @ w ---------------------
  const int wid  = tid >> 5;
  const int lane = tid & 31;
  const int mt   = wid & 3;                 // M-tile (16 rows of this 64-row slab)
  const int ng   = wid >> 2;                // N-group: tiles ng*4 .. ng*4+3
  const int lm   = lane & 15;
  const int kbA  = (lane < 16) ? 0 : 8;     // 16-bit A frag K sub-base
  const int kbB  = (lane < 16) ? 0 : 8;     // 16-bit B frag K sub-base

  v8f acc[2][4];
  #pragma unroll
  for (int a = 0; a < 2; ++a)
    #pragma unroll
    for (int j = 0; j < 4; ++j)
      acc[a][j] = (v8f){0.f,0.f,0.f,0.f,0.f,0.f,0.f,0.f};

  const size_t wbase = (size_t)f * 2 * Kd * Kd;
  #pragma unroll 1
  for (int a = 0; a < 2; ++a) {
    const float* wsrc = w + wbase + (size_t)a * Kd * Kd;   // [i][o], o fastest
    #pragma unroll 1
    for (int c = 0; c < 2; ++c) {
      const int i0 = c * 64;
      // stage transposed f16 weight chunk: sW[o][i] = w[i0+i][o]
      #pragma unroll 4
      for (int it = 0; it < 32; ++it) {
        int idx = it * 256 + tid;
        int o = idx & 127;
        int i = idx >> 7;
        sW[o][i] = (_Float16)wsrc[(size_t)(i0 + i) * Kd + o];
      }
      if (c == 0) __builtin_prefetch(wsrc + (size_t)64 * Kd, 0, 1);
      __syncthreads();
      #pragma unroll
      for (int kk = 0; kk < 64; kk += 32) {
        const int ka = i0 + kk;
        // A fragment: lanes 0-15 -> K kb..kb+7 & kb+16..kb+23 (kb=0), lanes 16-31 kb=8
        v8h alo = *(const v8h*)&sA[a][mt*16 + lm][ka + kbA];
        v8h ahi = *(const v8h*)&sA[a][mt*16 + lm][ka + 16 + kbA];
        v16h af = __builtin_shufflevector(alo, ahi,
                    0,1,2,3,4,5,6,7,8,9,10,11,12,13,14,15);
        #pragma unroll
        for (int j = 0; j < 4; ++j) {
          const int nt = ng*4 + j;
          // B fragment: same K order as A (kb..kb+7 & kb+16..kb+23, kb=kbB) at N=lane%16
          v8h blo = *(const v8h*)&sW[nt*16 + lm][kk + kbB];
          v8h bhi = *(const v8h*)&sW[nt*16 + lm][kk + kbB + 16];
          v16h bf = __builtin_shufflevector(blo, bhi,
                      0,1,2,3,4,5,6,7,8,9,10,11,12,13,14,15);
          acc[a][j] = __builtin_amdgcn_wmma_f32_16x16x32_f16(
              false, af, false, bf, (short)0, acc[a][j], false, false);
        }
      }
      __syncthreads();   // protect sW restage
    }
  }

  // ---- Epilogue: log(proj)+m summed over the 2 children, store f32 --------
  {
    const int rowoff = (lane >> 4) * 8;   // C/D layout: lanes 16-31 hold M=8..15
    #pragma unroll
    for (int j = 0; j < 4; ++j) {
      const int nt = ng*4 + j;
      #pragma unroll
      for (int v = 0; v < 8; ++v) {
        const int brow = mt*16 + rowoff + v;
        float r = __logf(acc[0][j][v]) + sM[0][brow]
                + __logf(acc[1][j][v]) + sM[1][brow];
        out[((size_t)f * Bsz + bb + brow) * Kd + nt*16 + lm] = r;
      }
    }
  }
}

// Final layer: F=2 -> 1, Ko=1. prev: (2, B, 128), w: (1, 2, 128, 1).
__global__ __launch_bounds__(256)
void pc_root_layer(const float* __restrict__ prev, const float* __restrict__ w,
                   float* __restrict__ out)
{
  const int b = threadIdx.x;
  float r = 0.f;
  #pragma unroll
  for (int a = 0; a < 2; ++a) {
    const float* row = prev + ((size_t)a * Bsz + b) * Kd;
    float m = -3.0e38f;
    #pragma unroll 8
    for (int k = 0; k < Kd; ++k) m = fmaxf(m, row[k]);
    float s = 0.f;
    #pragma unroll 8
    for (int k = 0; k < Kd; ++k) s += __expf(row[k] - m) * w[a * Kd + k];
    r += __logf(s) + m;
  }
  out[b] = r;
}

extern "C" void kernel_launch(void* const* d_in, const int* in_sizes, int n_in,
                              void* d_out, int out_size, void* d_ws, size_t ws_size,
                              hipStream_t stream) {
  (void)in_sizes; (void)n_in; (void)out_size; (void)ws_size;
  const float* x  = (const float*)d_in[0];
  const float* mu = (const float*)d_in[1];
  // d_in[2..11] = weights[0..9]

  float* buf0 = (float*)d_ws;                                   // up to (512,256,128) f32
  float* buf1 = (float*)((char*)d_ws + (size_t)512*Bsz*Kd*4);   // up to (256,256,128) f32

  // Layer 1: fused Gaussian input layer + CP layer (Fo = 512)
  pc_cp_layer<true><<<dim3(Bsz/BT, 512), 256, 0, stream>>>(
      x, mu, nullptr, (const float*)d_in[2], buf0);

  // Layers 2..9: generic CP layers (Fo = 256 ... 2), ping-pong buffers
  const float* cur = buf0;
  float* dst = buf1;
  int Fo = 256;
  for (int l = 1; l <= 8; ++l) {
    pc_cp_layer<false><<<dim3(Bsz/BT, Fo), 256, 0, stream>>>(
        nullptr, nullptr, cur, (const float*)d_in[2 + l], dst);
    cur = dst;
    dst = (dst == buf1) ? buf0 : buf1;
    Fo >>= 1;
  }

  // Layer 10: root (Fo = 1, Ko = 1) -> d_out (1, B)
  pc_root_layer<<<1, 256, 0, stream>>>(cur, (const float*)d_in[11], (float*)d_out);
}
